// AttentionEncoder_41961830482586
// MI455X (gfx1250) — compile-verified, matrix operand order checked by reading, not run on hardware
//
#include <hip/hip_runtime.h>
#include <hip/hip_bf16.h>

typedef __attribute__((ext_vector_type(16))) _Float16 v16h;
typedef __attribute__((ext_vector_type(8)))  _Float16 v8h;
typedef __attribute__((ext_vector_type(8)))  float    v8f;

#define S_ 512
#define D_ 128
#define C_ 128
#define B_ 128
#define O_ 256
#define NEGBIG -3.4e38f

// ---------------- WMMA fragment helper (ISA 7.12.2 layout, wave32) ----------------
// Both A and (pre-transposed) B operands are f16 row-major with the lane's 16
// halves = two contiguous 8-half runs: K in [k0+8*half, +8) and [k0+16+8*half, +8).
// => exactly two global_load_b128 per fragment, no conversions.
__device__ __forceinline__ v16h load_frag_rm(const _Float16* __restrict__ M, int ld,
                                             int r0, int k0, int lane) {
  int r = lane & 15, half = lane >> 4;
  const _Float16* base = M + (r0 + r) * ld + k0 + half * 8;
  v8h lo = *reinterpret_cast<const v8h*>(base);
  v8h hi = *reinterpret_cast<const v8h*>(base + 16);
  v16h f;
#pragma unroll
  for (int j = 0; j < 8; ++j) { f[j] = lo[j]; f[j + 8] = hi[j]; }
  return f;
}

// ---------------- pack: one-time f32->f16 convert (+ transpose for B operands) ------
// Regions: Qh[S*C] | cT[D*C] (= concept_emb^T) | WkT[D*D] | WmapT[O*D]
__global__ void pack_f16_kernel(const float* __restrict__ Q, const float* __restrict__ cemb,
                                const float* __restrict__ Wk, const float* __restrict__ Wmap,
                                _Float16* __restrict__ Qh, _Float16* __restrict__ cT,
                                _Float16* __restrict__ WkT, _Float16* __restrict__ WmapT) {
  int idx = blockIdx.x * 256 + threadIdx.x;        // 131072 threads
  if (idx < 65536) {                               // Qh: straight convert, coalesced
    Qh[idx] = (_Float16)Q[idx];
  } else if (idx < 65536 + 16384) {                // cT[d*C+c] = cemb[c*D+d]
    int i = idx - 65536, d = i >> 7, c = i & 127;
    cT[i] = (_Float16)cemb[c * D_ + d];
  } else if (idx < 65536 + 32768) {                // WkT[d*D+k] = Wk[k*D+d]
    int i = idx - (65536 + 16384), d = i >> 7, k = i & 127;
    WkT[i] = (_Float16)Wk[k * D_ + d];
  } else {                                         // WmapT[o*D+k] = Wmap[k*O+o]
    int i = idx - (65536 + 32768), o = i >> 7, k = i & 127;
    WmapT[i] = (_Float16)Wmap[k * O_ + o];
  }
}

// ---------------- small prep kernels ----------------
// sc[s] = exer_lam[s] / sum_c Q[s,c]   (folds the Rasch division out of the WMMA epilogue)
__global__ void scale_kernel(const float* __restrict__ Q, const float* __restrict__ elam,
                             float* __restrict__ sc) {
  int s = blockIdx.x * 256 + threadIdx.x;          // 512 threads total
  float acc = 0.f;
#pragma unroll 8
  for (int c = 0; c < C_; ++c) acc += Q[s * C_ + c];
  sc[s] = elam[s] / acc;                            // acc >= 1 by construction
}

// qr = resp_emb @ Wq + bq ; vr = resp_emb @ Wv + bv   (3 x 128 each)
__global__ void qrvr_kernel(const float* __restrict__ resp,
                            const float* __restrict__ Wq, const float* __restrict__ bq,
                            const float* __restrict__ Wv, const float* __restrict__ bv,
                            float* __restrict__ qr, float* __restrict__ vr) {
  int d = threadIdx.x;                              // 128 threads
  for (int r = 0; r < 3; ++r) {
    float aq = bq[d], av = bv[d];
    for (int k = 0; k < D_; ++k) {
      float x = resp[r * D_ + k];
      aq = fmaf(x, Wq[k * D_ + d], aq);
      av = fmaf(x, Wv[k * D_ + d], av);
    }
    qr[r * D_ + d] = aq;
    vr[r * D_ + d] = av;
  }
}

// exh = f16( exer_emb + sc[s] * (Q @ concept_emb) )   via WMMA
// grid: 64 blocks x 128 thr (4 waves => 4 tiles); 256 tiles of 16x16 over [512,128]
__global__ void ex_wmma_kernel(const _Float16* __restrict__ Qh, const _Float16* __restrict__ cT,
                               const float* __restrict__ eemb, const float* __restrict__ sc,
                               _Float16* __restrict__ exh) {
  int lane = threadIdx.x & 31;
  int tile = blockIdx.x * 4 + (threadIdx.x >> 5);   // 0..255
  int s0 = (tile >> 3) * 16;                        // 32 s-tiles
  int d0 = (tile & 7) * 16;                         // 8 d-tiles
  __builtin_prefetch(cT + d0 * C_, 0, 3);           // global_prefetch_b8
  v8f acc = {};
#pragma unroll
  for (int k0 = 0; k0 < C_; k0 += 32) {
    v16h a = load_frag_rm(Qh, C_, s0, k0, lane);    // A: rows of Q
    v16h b = load_frag_rm(cT, C_, d0, k0, lane);    // B: rows of concept^T = columns
    acc = __builtin_amdgcn_wmma_f32_16x16x32_f16(false, a, false, b, (short)0, acc, false, false);
  }
  int n = lane & 15, hi = lane >> 4;
#pragma unroll
  for (int i = 0; i < 8; ++i) {
    int s = s0 + i + 8 * hi, d = d0 + n;
    exh[s * D_ + d] = (_Float16)fmaf(sc[s], acc[i], eemb[s * D_ + d]);
  }
}

// kbuf[512,128] = exh @ Wk + bk   via WMMA (f32 result feeds the logit dots)
__global__ void proj_wmma_kernel(const _Float16* __restrict__ exh, const _Float16* __restrict__ WkT,
                                 const float* __restrict__ bias, float* __restrict__ Y) {
  int lane = threadIdx.x & 31;
  int tile = blockIdx.x * 4 + (threadIdx.x >> 5);   // 0..255
  int s0 = (tile >> 3) * 16;
  int d0 = (tile & 7) * 16;
  __builtin_prefetch(WkT + d0 * D_, 0, 3);
  v8f acc = {};
#pragma unroll
  for (int k0 = 0; k0 < D_; k0 += 32) {
    v16h a = load_frag_rm(exh, D_, s0, k0, lane);
    v16h b = load_frag_rm(WkT, D_, d0, k0, lane);
    acc = __builtin_amdgcn_wmma_f32_16x16x32_f16(false, a, false, b, (short)0, acc, false, false);
  }
  int n = lane & 15, hi = lane >> 4;
  float bn = bias[d0 + n];
#pragma unroll
  for (int i = 0; i < 8; ++i)
    Y[(s0 + i + 8 * hi) * D_ + d0 + n] = acc[i] + bn;
}

// G[r,h,t] = (1/sqrt(32)) * dot(qr[r+1, h*32:(h+1)*32], k[t, h*32:(h+1)*32])  -> [2,4,512]
__global__ void g_kernel(const float* __restrict__ qr, const float* __restrict__ kbuf,
                         float* __restrict__ G) {
  int idx = blockIdx.x * 256 + threadIdx.x;         // 4096 total
  int r = idx >> 11;
  int h = (idx >> 9) & 3;
  int t = idx & (S_ - 1);
  const float* qrow = qr + (r + 1) * D_ + h * 32;
  const float* krow = kbuf + t * D_ + h * 32;
  float acc = 0.f;
#pragma unroll
  for (int j = 0; j < 32; ++j) acc = fmaf(qrow[j], krow[j], acc);
  G[idx] = acc * 0.17677669529663687f;              // 1/sqrt(32)
}

// Per batch: class counts, per-(r,h) masked softmax class-mass, pooled[b,:] (exact collapse)
// grid: 128 blocks x 256 thr. wave wid handles combo (r = wid>>2, h = wid&3).
// Writes pooled directly as f16 (it only feeds the final WMMA A-operand).
__global__ void softmax_pool_kernel(const int* __restrict__ p, const float* __restrict__ G,
                                    const float* __restrict__ vr, _Float16* __restrict__ pooledh) {
  int b = blockIdx.x;
  int tid = threadIdx.x;
  int lane = tid & 31, wid = tid >> 5;
  __shared__ int   sn[2];
  __shared__ float sA[8][2];
  __shared__ float coef[4][2];
  if (tid < 2) sn[tid] = 0;
  __syncthreads();

  // class counts n1, n2
  int c1 = 0, c2 = 0;
  for (int t = tid; t < S_; t += 256) {
    int pv = p[b * S_ + t];
    c1 += (pv == 1);
    c2 += (pv == 2);
  }
  for (int off = 16; off; off >>= 1) { c1 += __shfl_xor(c1, off); c2 += __shfl_xor(c2, off); }
  if (lane == 0) { atomicAdd(&sn[0], c1); atomicAdd(&sn[1], c2); }

  // masked max + class-partitioned exp sums for this wave's (r,h)
  int r = wid >> 2, h = wid & 3;
  const float* Gr = G + r * 2048 + h * 512;
  float gv[16]; int pvv[16];
  float m = NEGBIG;
#pragma unroll
  for (int i = 0; i < 16; ++i) {
    int t = lane + 32 * i;
    int pv = p[b * S_ + t];
    float g = (pv != 0) ? Gr[t] : NEGBIG;
    pvv[i] = pv; gv[i] = g;
    m = fmaxf(m, g);
  }
  for (int off = 16; off; off >>= 1) m = fmaxf(m, __shfl_xor(m, off));
  float s1 = 0.f, s2 = 0.f;
#pragma unroll
  for (int i = 0; i < 16; ++i) {
    if (pvv[i] != 0) {
      float e = __expf(gv[i] - m);
      if (pvv[i] == 1) s1 += e; else s2 += e;
    }
  }
  for (int off = 16; off; off >>= 1) { s1 += __shfl_xor(s1, off); s2 += __shfl_xor(s2, off); }
  if (lane == 0) {
    float tot = s1 + s2;
    float inv = (tot > 0.f) ? 1.f / tot : 0.f;
    sA[wid][0] = s1 * inv;
    sA[wid][1] = s2 * inv;
  }
  __syncthreads();

  int n1 = sn[0], n2 = sn[1], N = n1 + n2;
  if (tid < 8) {
    int hh = tid >> 1, cc = tid & 1;
    float fN = (N > 0) ? 1.f / (float)N : 0.f;
    coef[hh][cc] = ((float)n1 * sA[hh][cc] + (float)n2 * sA[4 + hh][cc]) * fN;
  }
  __syncthreads();
  if (tid < D_) {
    int d = tid, hh = d >> 5;
    float val = (N > 0) ? (coef[hh][0] * vr[1 * D_ + d] + coef[hh][1] * vr[2 * D_ + d]) : 0.f;
    pooledh[b * D_ + d] = (_Float16)val;
  }
}

// out = sigmoid(pooled[128,128] @ Wmap[128,256] + bmap)  via WMMA
// grid: 32 blocks x 128 thr (4 waves); 128 tiles (8 m-tiles x 16 n-tiles)
__global__ void map_wmma_kernel(const _Float16* __restrict__ pooledh,
                                const _Float16* __restrict__ WmapT,
                                const float* __restrict__ bmap, float* __restrict__ out) {
  int lane = threadIdx.x & 31;
  int tile = blockIdx.x * 4 + (threadIdx.x >> 5);   // 0..127
  int m0 = (tile >> 4) * 16;                        // 8 B-tiles
  int n0 = (tile & 15) * 16;                        // 16 O-tiles
  __builtin_prefetch(WmapT + n0 * D_, 0, 3);
  v8f acc = {};
#pragma unroll
  for (int k0 = 0; k0 < D_; k0 += 32) {
    v16h a = load_frag_rm(pooledh, D_, m0, k0, lane);
    v16h b = load_frag_rm(WmapT, D_, n0, k0, lane);
    acc = __builtin_amdgcn_wmma_f32_16x16x32_f16(false, a, false, b, (short)0, acc, false, false);
  }
  int n = lane & 15, hi = lane >> 4;
  float bn = bmap[n0 + n];
#pragma unroll
  for (int i = 0; i < 8; ++i) {
    float x = acc[i] + bn;
    out[(m0 + i + 8 * hi) * O_ + n0 + n] = 1.f / (1.f + __expf(-x));
  }
}

extern "C" void kernel_launch(void* const* d_in, const int* in_sizes, int n_in,
                              void* d_out, int out_size, void* d_ws, size_t ws_size,
                              hipStream_t stream) {
  const int*   p     = (const int*)  d_in[0];   // [B,S]
  const float* eemb  = (const float*)d_in[1];   // [S,D]
  const float* elam  = (const float*)d_in[2];   // [S,1]
  const float* cemb  = (const float*)d_in[3];   // [C,D]
  const float* resp  = (const float*)d_in[4];   // [3,D]
  const float* Q     = (const float*)d_in[5];   // [S,C]
  const float* Wq    = (const float*)d_in[6];
  const float* bq    = (const float*)d_in[7];
  const float* Wk    = (const float*)d_in[8];
  const float* bk    = (const float*)d_in[9];
  const float* Wv    = (const float*)d_in[10];
  const float* bv    = (const float*)d_in[11];
  const float* Wmap  = (const float*)d_in[12];  // [D,O]
  const float* bmap  = (const float*)d_in[13];
  float* out = (float*)d_out;                   // [B,O] f32

  // workspace layout: f32 region, then 16B-aligned f16 region
  float* w      = (float*)d_ws;
  float* kbuf   = w;                  // 512*128 = 65536 f
  float* qr     = kbuf + 65536;       // 512 f (3*128 used)
  float* vr     = qr   + 512;         // 512 f
  float* sc     = vr   + 512;         // 512 f  (elam/qsum)
  float* G      = sc   + 512;         // 2*4*512 = 4096 f
  _Float16* hbase  = (_Float16*)(G + 4096);      // f32-aligned => 16B-aligned offsets below
  _Float16* Qh     = hbase;           // 512*128
  _Float16* cT     = Qh     + 65536;  // 128*128 (concept_emb^T)
  _Float16* WkT    = cT     + 16384;  // 128*128 (Wk^T)
  _Float16* WmapT  = WkT    + 16384;  // 256*128 (Wmap^T)
  _Float16* exh    = WmapT  + 32768;  // 512*128
  _Float16* pooledh= exh    + 65536;  // 128*128

  pack_f16_kernel<<<512, 256, 0, stream>>>(Q, cemb, Wk, Wmap, Qh, cT, WkT, WmapT);
  scale_kernel<<<2, 256, 0, stream>>>(Q, elam, sc);
  qrvr_kernel<<<1, 128, 0, stream>>>(resp, Wq, bq, Wv, bv, qr, vr);
  ex_wmma_kernel<<<64, 128, 0, stream>>>(Qh, cT, eemb, sc, exh);
  proj_wmma_kernel<<<64, 128, 0, stream>>>(exh, WkT, bk, kbuf);
  g_kernel<<<16, 256, 0, stream>>>(qr, kbuf, G);
  softmax_pool_kernel<<<128, 256, 0, stream>>>(p, G, vr, pooledh);
  map_wmma_kernel<<<32, 128, 0, stream>>>(pooledh, WmapT, bmap, out);
}
